// TransformerEncoderLayer_30958124270274
// MI455X (gfx1250) — matrix operand order repaired and checked by reading, not compiled, not run on hardware
//
#include <hip/hip_runtime.h>
#include <hip/hip_bf16.h>

// ---------------------------------------------------------------------------
// Types for CDNA5 WMMA
// ---------------------------------------------------------------------------
typedef __attribute__((ext_vector_type(16))) __bf16 v16bf;
typedef __attribute__((ext_vector_type(8)))  float  v8f;
typedef int v4i __attribute__((vector_size(16)));   // v4i32 for async-LDS builtin

__device__ __forceinline__ unsigned short f2bf(float f) {
    unsigned int u = __float_as_uint(f);
    u = (u + 0x7FFFu + ((u >> 16) & 1u)) >> 16;   // round-to-nearest-even
    return (unsigned short)u;
}

__device__ __forceinline__ void store_out(float* p, float v) { *p = v; }
__device__ __forceinline__ void store_out(unsigned short* p, float v) { *p = f2bf(v); }

// ---------------------------------------------------------------------------
// CDNA5 async global->LDS staging (bypasses VGPRs, tracked by ASYNCcnt).
// Falls back to synchronous b128 copy if the toolchain lacks the builtins.
// ---------------------------------------------------------------------------
#if __has_builtin(__builtin_amdgcn_global_load_async_to_lds_b128) && \
    __has_builtin(__builtin_amdgcn_s_wait_asynccnt)
#define GEMM_ASYNC_LDS 1
#else
#define GEMM_ASYNC_LDS 0
#endif

__device__ __forceinline__ void copy16_g2l(unsigned short* lds_dst,
                                           const unsigned short* g_src) {
#if GEMM_ASYNC_LDS
    __builtin_amdgcn_global_load_async_to_lds_b128(
        (__attribute__((address_space(1))) v4i*)g_src,
        (__attribute__((address_space(3))) v4i*)lds_dst,
        0, 0);
#else
    *(float4*)lds_dst = *(const float4*)g_src;
#endif
}

template <int N>
__device__ __forceinline__ void wait_async() {
#if GEMM_ASYNC_LDS
    __builtin_amdgcn_s_wait_asynccnt(N);
#endif
}

// ---------------------------------------------------------------------------
// LayerNorm (f32 in) -> bf16 row-major out, optional bf16 transposed out
// one 256-thread block per row of D=1024
// ---------------------------------------------------------------------------
__global__ __launch_bounds__(256)
void layernorm_bf16_kernel(const float* __restrict__ x,
                           const float* __restrict__ g,
                           const float* __restrict__ bta,
                           unsigned short* __restrict__ out,
                           unsigned short* __restrict__ outT,
                           int D, int S)
{
    const int row = blockIdx.x;
    const int tid = threadIdx.x;
    const float* xr = x + (long long)row * D;

    float v[4];
    float s = 0.f, ss = 0.f;
#pragma unroll
    for (int i = 0; i < 4; ++i) {
        v[i] = xr[tid + i * 256];
        s += v[i];
        ss += v[i] * v[i];
    }
    __shared__ float2 red[256];
    red[tid] = make_float2(s, ss);
    __syncthreads();
    for (int off = 128; off > 0; off >>= 1) {
        if (tid < off) {
            float2 o = red[tid + off];
            red[tid].x += o.x;
            red[tid].y += o.y;
        }
        __syncthreads();
    }
    const float mean = red[0].x / (float)D;
    const float var  = red[0].y / (float)D - mean * mean;
    const float rstd = rsqrtf(var + 1e-5f);

    const int batch = row / S;
    const int q     = row % S;
#pragma unroll
    for (int i = 0; i < 4; ++i) {
        const int d = tid + i * 256;
        const float y = (v[i] - mean) * rstd * g[d] + bta[d];
        const unsigned short h = f2bf(y);
        out[(long long)row * D + d] = h;
        if (outT)
            outT[(long long)batch * D * S + (long long)d * S + q] = h;
    }
}

// ---------------------------------------------------------------------------
// f32 [R x C] row-major  ->  bf16 [C x R] row-major (convert + transpose)
// block 32x8, tile 32x32
// ---------------------------------------------------------------------------
__global__ __launch_bounds__(256)
void transpose_f32_to_bf16_kernel(const float* __restrict__ in,
                                  unsigned short* __restrict__ outT,
                                  int R, int C)
{
    __shared__ float t[32][33];
    const int tx = threadIdx.x, ty = threadIdx.y;
    const int c0 = blockIdx.x * 32, r0 = blockIdx.y * 32;
#pragma unroll
    for (int i = 0; i < 32; i += 8)
        t[ty + i][tx] = in[(long long)(r0 + ty + i) * C + (c0 + tx)];
    __syncthreads();
#pragma unroll
    for (int i = 0; i < 32; i += 8)
        outT[(long long)(c0 + ty + i) * R + (r0 + tx)] = f2bf(t[tx][ty + i]);
}

// ---------------------------------------------------------------------------
// Row softmax: f32 scores [rows x S] -> bf16 probs, block per row, S = 2048
// ---------------------------------------------------------------------------
__global__ __launch_bounds__(256)
void softmax_bf16_kernel(const float* __restrict__ sc,
                         unsigned short* __restrict__ pr,
                         int Scols)
{
    const long long row = blockIdx.x;
    const float* s = sc + row * Scols;
    unsigned short* p = pr + row * Scols;
    const int tid = threadIdx.x;

    float v[8];
    float mx = -3.4e38f;
#pragma unroll
    for (int i = 0; i < 8; ++i) {
        v[i] = s[tid + i * 256];
        mx = fmaxf(mx, v[i]);
    }
    __shared__ float red[256];
    red[tid] = mx;
    __syncthreads();
    for (int off = 128; off > 0; off >>= 1) {
        if (tid < off) red[tid] = fmaxf(red[tid], red[tid + off]);
        __syncthreads();
    }
    mx = red[0];
    __syncthreads();

    float sum = 0.f;
#pragma unroll
    for (int i = 0; i < 8; ++i) {
        v[i] = __expf(v[i] - mx);
        sum += v[i];
    }
    red[tid] = sum;
    __syncthreads();
    for (int off = 128; off > 0; off >>= 1) {
        if (tid < off) red[tid] += red[tid + off];
        __syncthreads();
    }
    const float inv = 1.0f / red[0];
#pragma unroll
    for (int i = 0; i < 8; ++i)
        p[tid + i * 256] = f2bf(v[i] * inv);
}

// ---------------------------------------------------------------------------
// NT GEMM:  C[M x N] = epilogue( scale * A(bf16, MxK rm) * B(bf16, NxK rm)^T )
//
// Block = 256 threads = 8 waves.  Block tile 128x128; wave grid 4x2, each wave
// owns a 32x64 tile as 2x4 v_wmma_f32_16x16x32_bf16 accumulators.  K is staged
// through double-buffered LDS panels of width 64, filled with CDNA5 async
// global->LDS b128 transfers overlapped with WMMA compute.
// All of M, N are multiples of 128 and K a multiple of 64 here (no edges).
// Epilogues: 0=none(f32), 1=+residual(f32), 2=+bias,GELU(bf16), 3=+bias+resid(f32)
// ---------------------------------------------------------------------------
#define GEMM_TM   128   // block tile M
#define GEMM_TN   128   // block tile N
#define GEMM_KC   64    // K panel width
#define GEMM_LDT  72    // padded LDS leading dim (72*2B = 144B, 16B-aligned rows)

template <int EPI, typename OutT>
__global__ __launch_bounds__(256)
void gemm_nt_kernel(const unsigned short* __restrict__ A,
                    const unsigned short* __restrict__ B,
                    OutT* __restrict__ C,
                    int M, int N, int K,
                    long long sA, long long sB, long long sC,
                    float scale,
                    const float* __restrict__ bias,
                    const float* __restrict__ resid, long long sR)
{
    __shared__ __align__(16) unsigned short As[2][GEMM_TM][GEMM_LDT];
    __shared__ __align__(16) unsigned short Bs[2][GEMM_TN][GEMM_LDT];

    const int z = blockIdx.z;
    A += z * sA;
    B += z * sB;
    C += z * sC;
    const float* R = resid ? resid + z * sR : nullptr;

    const int tid  = threadIdx.x;
    const int wave = tid >> 5;
    const int lane = tid & 31;
    const int half = lane >> 4;   // selects K sub-chunk per ISA A/B layouts
    const int lr   = lane & 15;

    const int wrow = (wave >> 1) * 32;  // wave tile origin inside 128x128
    const int wcol = (wave & 1) * 64;

    const int m0 = blockIdx.y * GEMM_TM;
    const int n0 = blockIdx.x * GEMM_TN;

    v8f acc[2][4];
#pragma unroll
    for (int i = 0; i < 2; ++i)
#pragma unroll
        for (int j = 0; j < 4; ++j)
            acc[i][j] = (v8f)(0.0f);

    // Stage one 128x64 panel of A and of B into LDS buffer s.
    // 2048 16-byte chunks / 256 threads -> 8 async b128 ops per thread.
    auto stage_load = [&](int s, int kt) {
#pragma unroll
        for (int i = 0; i < 4; ++i) {
            const int c  = tid + i * 256;   // 1024 chunks per matrix
            const int r  = c >> 3;          // 0..127
            const int c8 = (c & 7) * 8;     // 0,8,...,56
            copy16_g2l(&As[s][r][c8], &A[(long long)(m0 + r) * K + kt + c8]);
            copy16_g2l(&Bs[s][r][c8], &B[(long long)(n0 + r) * K + kt + c8]);
        }
    };

    const int NK = K / GEMM_KC;
    stage_load(0, 0);

    for (int kt = 0; kt < NK; ++kt) {
        const int cur = kt & 1;
        if (kt + 1 < NK) {
            stage_load(cur ^ 1, (kt + 1) * GEMM_KC);  // prefetch next panel
            wait_async<8>();                          // this panel's 8 ops done
        } else {
            wait_async<0>();
        }
        __syncthreads();

#pragma unroll
        for (int ks = 0; ks < GEMM_KC; ks += 32) {
            v16bf a[2], b[4];
#pragma unroll
            for (int i = 0; i < 2; ++i) {
                // A 16x32 bf16 fragment: lane = row%16;
                // K chunks {half*8..} and {16+half*8..}
                const int ar = wrow + i * 16 + lr;
                ((float4*)&a[i])[0] = *(const float4*)&As[cur][ar][ks + half * 8];
                ((float4*)&a[i])[1] = *(const float4*)&As[cur][ar][ks + 16 + half * 8];
            }
#pragma unroll
            for (int j = 0; j < 4; ++j) {
                // B 32x16 bf16 fragment: lane = col%16;
                // K chunks {half*8..} and {16+half*8..} (Bs holds B^T rows = C columns)
                const int br = wcol + j * 16 + lr;
                ((float4*)&b[j])[0] = *(const float4*)&Bs[cur][br][ks + half * 8];
                ((float4*)&b[j])[1] = *(const float4*)&Bs[cur][br][ks + 16 + half * 8];
            }
#pragma unroll
            for (int i = 0; i < 2; ++i)
#pragma unroll
                for (int j = 0; j < 4; ++j)
                    acc[i][j] = __builtin_amdgcn_wmma_f32_16x16x32_bf16(
                        false, a[i], false, b[j], (short)0, acc[i][j],
                        false, false);
        }
        __syncthreads();
    }

    // Epilogue + store. C/D layout: VGPR r -> row = r + 8*(lane>=16), col = lane&15.
#pragma unroll
    for (int i = 0; i < 2; ++i) {
#pragma unroll
        for (int j = 0; j < 4; ++j) {
            const int cn = n0 + wcol + j * 16 + lr;
#pragma unroll
            for (int r = 0; r < 8; ++r) {
                const int cm = m0 + wrow + i * 16 + half * 8 + r;
                float v = acc[i][j][r] * scale;
                if (EPI == 1) {                       // + residual
                    v += R[(long long)cm * N + cn];
                } else if (EPI == 2) {                // + bias, exact GELU
                    v += bias[cn];
                    v = 0.5f * v * (1.0f + erff(v * 0.70710678118654752f));
                } else if (EPI == 3) {                // + bias + residual
                    v += bias[cn] + R[(long long)cm * N + cn];
                }
                store_out(&C[(long long)cm * N + cn], v);
            }
        }
    }
}

// ---------------------------------------------------------------------------
// Host-side orchestration
// ---------------------------------------------------------------------------
extern "C" void kernel_launch(void* const* d_in, const int* in_sizes, int n_in,
                              void* d_out, int out_size, void* d_ws, size_t ws_size,
                              hipStream_t stream)
{
    const float* src    = (const float*)d_in[0];
    const float* gamma1 = (const float*)d_in[1];
    const float* beta1  = (const float*)d_in[2];
    const float* gamma2 = (const float*)d_in[3];
    const float* beta2  = (const float*)d_in[4];
    const float* w1     = (const float*)d_in[5];
    const float* b1     = (const float*)d_in[6];
    const float* w2     = (const float*)d_in[7];
    const float* b2     = (const float*)d_in[8];
    float* out = (float*)d_out;

    constexpr int B = 4, S = 2048, D = 1024, F = 4096;
    constexpr long long BS = (long long)B * S;

    char* p = (char*)d_ws;
    auto alloc = [&](long long bytes) -> void* {
        void* r = (void*)p;
        p += (bytes + 255) & ~255ll;
        return r;
    };
    unsigned short* normx  = (unsigned short*)alloc(BS * D * 2);          // LN1 out (bf16)
    unsigned short* normxT = (unsigned short*)alloc(BS * D * 2);          // per-batch (D x S)
    float*          scores = (float*)alloc((long long)B * S * S * 4);     // QK^T / sqrt(D)
    unsigned short* probs  = (unsigned short*)alloc((long long)B * S * S * 2);
    float*          xbuf   = (float*)alloc(BS * D * 4);                   // src + attn
    unsigned short* hbuf   = (unsigned short*)alloc(BS * D * 2);          // LN2 out (bf16)
    unsigned short* act    = (unsigned short*)alloc(BS * F * 2);          // gelu(h@w1+b1)
    unsigned short* w1T    = (unsigned short*)alloc((long long)D * F * 2);
    unsigned short* w2T    = (unsigned short*)alloc((long long)D * F * 2);

    // 1) LN1 -> normx (bf16) and normxT (bf16, per-batch D x S)
    layernorm_bf16_kernel<<<(int)BS, 256, 0, stream>>>(
        src, gamma1, beta1, normx, normxT, D, S);

    // 2) Weight conversion+transpose to NT form
    transpose_f32_to_bf16_kernel<<<dim3(F / 32, D / 32), dim3(32, 8), 0, stream>>>(
        w1, w1T, D, F);
    transpose_f32_to_bf16_kernel<<<dim3(D / 32, F / 32), dim3(32, 8), 0, stream>>>(
        w2, w2T, F, D);

    // 3) scores[b] = (normx[b] @ normx[b]^T) / 32   (S x S, f32)
    gemm_nt_kernel<0, float><<<dim3(S / 128, S / 128, B), 256, 0, stream>>>(
        normx, normx, scores, S, S, D,
        (long long)S * D, (long long)S * D, (long long)S * S,
        1.0f / 32.0f, nullptr, nullptr, 0);

    // 4) probs = softmax(scores) -> bf16
    softmax_bf16_kernel<<<(int)(B * (long long)S), 256, 0, stream>>>(scores, probs, S);

    // 5) x[b] = src[b] + probs[b] @ (normxT[b])^T   (S x D, f32)
    gemm_nt_kernel<1, float><<<dim3(D / 128, S / 128, B), 256, 0, stream>>>(
        probs, normxT, xbuf, S, D, S,
        (long long)S * S, (long long)D * S, (long long)S * D,
        1.0f, nullptr, src, (long long)S * D);

    // 6) LN2 -> h (bf16)
    layernorm_bf16_kernel<<<(int)BS, 256, 0, stream>>>(
        xbuf, gamma2, beta2, hbuf, nullptr, D, S);

    // 7) act = gelu(h @ w1 + b1)  (BS x F, bf16)
    gemm_nt_kernel<2, unsigned short><<<dim3(F / 128, (int)(BS / 128), 1), 256, 0, stream>>>(
        hbuf, w1T, act, (int)BS, F, D,
        0, 0, 0, 1.0f, b1, nullptr, 0);

    // 8) out = x + act @ w2 + b2  (BS x D, f32)
    gemm_nt_kernel<3, float><<<dim3(D / 128, (int)(BS / 128), 1), 256, 0, stream>>>(
        act, w2T, out, (int)BS, D, F,
        0, 0, 0, 1.0f, b2, xbuf, 0);
}
